// RPC_20710332301412
// MI455X (gfx1250) — matrix operand order repaired and checked by reading, not compiled, not run on hardware
//
#include <hip/hip_runtime.h>

// ---------------------------------------------------------------------------
// GDANet forward on gfx1250 (MI455X).  wave32, WMMA f16->f32 for all GEMMs.
// GEMM uses cooperative LDS staging + ds_load_b128 fragment reads.
// ---------------------------------------------------------------------------

typedef __attribute__((ext_vector_type(16))) _Float16 v16h;
typedef __attribute__((ext_vector_type(8)))  _Float16 v8h;
typedef __attribute__((ext_vector_type(8)))  float    v8f;

#define B_SZ   16
#define N_PTS  1024
#define K_LOC  30
#define M_GDM  256
#define K_GDM  64

static __device__ __forceinline__ float bn_rs() { return rsqrtf(1.0f + 1e-5f); }

// ---------------------------------------------------------------------------
// Generic batched strided GEMM:  C = act( bn( alpha*A*B + bias ) ) + R
//   a(m,k) = A[m*Ars + k*Acs + b*Abs]   (element strides)
//   b(k,n) = B[k*Brs + n*Bcs + b*Bbs]
//   c(m,n) = C[m*Crs + n*Ccs + b*Cbs]
// Block = 128 threads (4 waves).  Block tile = 16(M) x 64(N); each wave owns
// one 16x16 WMMA tile.  A/B slabs staged through LDS in f16.
// ---------------------------------------------------------------------------
struct GemmDesc {
  const float* A; long Ars, Acs, Abs;
  const float* B; long Brs, Bcs, Bbs;
  float*       C; long Crs, Ccs, Cbs;
  const float* bias;                   // per-row m, may be null
  const float* bng; const float* bnb;  // batchnorm gamma/beta (eval, var=1), null
  const float* R; long Rrs, Rcs, Rbs;  // residual, null
  int M, N, K, act;                    // act: 0 none, 1 relu, 2 leaky(0.2)
  float alpha;
};

__global__ void gemm_wmma_kernel(GemmDesc d) {
  __shared__ _Float16 lds_a[16 * 32];   // A tile, m-major: [m*32 + k]
  __shared__ _Float16 lds_b[64 * 32];   // B tile, n-major: [nl*32 + k]

  const int tid  = threadIdx.x;
  const int wave = tid >> 5;
  const int lane = tid & 31;
  const int tm     = blockIdx.x << 4;
  const int tn_blk = blockIdx.y << 6;
  const int tn     = tn_blk + (wave << 4);
  const int b = blockIdx.z;
  const float* A  = d.A + (long)b * d.Abs;
  const float* Bp = d.B + (long)b * d.Bbs;
  const int row = lane & 15;   // A row / B col owned by this lane
  const int lg  = lane >> 4;   // lane group

  v8f acc = {};
  for (int k0 = 0; k0 < d.K; k0 += 32) {
    // ---- stage A slab (16x32), 4 elements/thread, branchless ----
#pragma unroll
    for (int e = 0; e < 4; ++e) {
      const int idx = tid + (e << 7);          // 0..511
      const int m = idx >> 5, k = idx & 31;
      const int gm = tm + m, gk = k0 + k;
      const bool v = (gm < d.M) && (gk < d.K);
      const long ai = (long)(v ? gm : 0) * d.Ars + (long)(v ? gk : 0) * d.Acs;
      const float av = A[ai];
      lds_a[(m << 5) + k] = (_Float16)(v ? av : 0.0f);
    }
    // ---- stage B slab (32x64), 16 elements/thread, branchless ----
#pragma unroll
    for (int e = 0; e < 16; ++e) {
      const int idx = tid + (e << 7);          // 0..2047
      const int nl = idx & 63, k = idx >> 6;
      const int gn = tn_blk + nl, gk = k0 + k;
      const bool v = (gn < d.N) && (gk < d.K);
      const long bi = (long)(v ? gk : 0) * d.Brs + (long)(v ? gn : 0) * d.Bcs;
      const float bv = Bp[bi];
      lds_b[(nl << 5) + k] = (_Float16)(v ? bv : 0.0f);
    }
    if (k0 + 32 < d.K) {   // speculative global_prefetch of next K slab
      __builtin_prefetch(&A[(long)(tm + (tid >> 5)) * d.Ars +
                            (long)(k0 + 32 + (tid & 31)) * d.Acs], 0, 1);
      __builtin_prefetch(&Bp[(long)(k0 + 32 + (tid >> 6)) * d.Brs +
                             (long)(tn_blk + (tid & 63)) * d.Bcs], 0, 1);
    }
    __syncthreads();
    // ---- fragments: 2x ds_load_b128 each (ISA 16-bit WMMA layouts) ----
    // A: af[i] = A16[row][(i&7) + 16*(i>>3) + 8*lg]  -> two contiguous chunks
    const v8h a0 = *(const v8h*)&lds_a[(row << 5) + (lg << 3)];
    const v8h a1 = *(const v8h*)&lds_a[(row << 5) + 16 + (lg << 3)];
    // B: bf[i] = B16[(i&7) + 16*(i>>3) + 8*lg][tn + row] -> two contiguous chunks
    const int nl32 = ((wave << 4) + row) << 5;
    const v8h b0 = *(const v8h*)&lds_b[nl32 + (lg << 3)];
    const v8h b1 = *(const v8h*)&lds_b[nl32 + 16 + (lg << 3)];
    const v16h af = __builtin_shufflevector(a0, a1, 0, 1, 2, 3, 4, 5, 6, 7,
                                            8, 9, 10, 11, 12, 13, 14, 15);
    const v16h bf = __builtin_shufflevector(b0, b1, 0, 1, 2, 3, 4, 5, 6, 7,
                                            8, 9, 10, 11, 12, 13, 14, 15);
    acc = __builtin_amdgcn_wmma_f32_16x16x32_f16(false, af, false, bf,
                                                 (short)0, acc, false, false);
    __syncthreads();
  }

  // ---- epilogue ----
  float* C = d.C + (long)b * d.Cbs;
  const float rs = bn_rs();
#pragma unroll
  for (int v = 0; v < 8; ++v) {
    const int m = tm + v + (lg << 3);
    const int n = tn + row;
    if (m >= d.M || n >= d.N) continue;
    float c = acc[v] * d.alpha;
    if (d.bias) c += d.bias[m];
    if (d.bng)  c = c * (d.bng[m] * rs) + d.bnb[m];
    if (d.act == 1)      c = fmaxf(c, 0.0f);
    else if (d.act == 2) c = (c > 0.0f) ? c : 0.2f * c;
    if (d.R) c += d.R[(long)b * d.Rbs + (long)m * d.Rrs + (long)n * d.Rcs];
    C[(long)m * d.Crs + (long)n * d.Ccs] = c;
  }
}

// ---------------------------------------------------------------------------
// KNN on raw xyz (k=30, self included, smallest sq-dist, low-index tiebreak)
// ---------------------------------------------------------------------------
__global__ void knn3_kernel(const float* __restrict__ x, int* __restrict__ idx) {
  const int t = blockIdx.x * blockDim.x + threadIdx.x;
  if (t >= B_SZ * N_PTS) return;
  const int b = t >> 10, n = t & 1023;
  const float* xb = x + (long)b * N_PTS * 3;
  const float px = xb[n * 3 + 0], py = xb[n * 3 + 1], pz = xb[n * 3 + 2];
  float bd[K_LOC]; int bi[K_LOC];
#pragma unroll
  for (int i = 0; i < K_LOC; ++i) { bd[i] = 3.0e38f; bi[i] = 0; }
  for (int m = 0; m < N_PTS; ++m) {
    const float dx = xb[m * 3 + 0] - px;
    const float dy = xb[m * 3 + 1] - py;
    const float dz = xb[m * 3 + 2] - pz;
    const float dsq = dx * dx + dy * dy + dz * dz;
    if (dsq < bd[K_LOC - 1]) {
      int p = K_LOC - 1;
      while (p > 0 && bd[p - 1] > dsq) { bd[p] = bd[p - 1]; bi[p] = bi[p - 1]; --p; }
      bd[p] = dsq; bi[p] = m;
    }
  }
  int* o = idx + (long)t * K_LOC;
  for (int i = 0; i < K_LOC; ++i) o[i] = bi[i];
}

// ---------------------------------------------------------------------------
// Local operator: gather neighbors -> feat(6,30) -> conv1(64)+bn+relu ->
// conv11(128)+bn+relu -> max over k.  One block per (b,n), 128 threads.
// ---------------------------------------------------------------------------
__global__ void local_op_kernel(const float* __restrict__ x, const int* __restrict__ idx,
                                const float* __restrict__ w1,  const float* __restrict__ b1,
                                const float* __restrict__ g1,  const float* __restrict__ be1,
                                const float* __restrict__ w11, const float* __restrict__ b11,
                                const float* __restrict__ g11, const float* __restrict__ be11,
                                float* __restrict__ x1) {
  __shared__ float sfeat[6][K_LOC];
  __shared__ float sh1[64][K_LOC];
  const int t = blockIdx.x;
  const int b = t >> 10, n = t & 1023;
  const int tid = threadIdx.x;
  const float* xb = x + (long)b * N_PTS * 3;
  if (tid < K_LOC) {
    const int m = idx[(long)t * K_LOC + tid];
    const float nx = xb[m * 3 + 0], ny = xb[m * 3 + 1], nz = xb[m * 3 + 2];
    sfeat[0][tid] = nx - xb[n * 3 + 0];
    sfeat[1][tid] = ny - xb[n * 3 + 1];
    sfeat[2][tid] = nz - xb[n * 3 + 2];
    sfeat[3][tid] = nx; sfeat[4][tid] = ny; sfeat[5][tid] = nz;
  }
  __syncthreads();
  const float rs = bn_rs();
  for (int e = tid; e < 64 * K_LOC; e += 128) {
    const int o = e / K_LOC, kk = e - o * K_LOC;
    float s = b1[o];
#pragma unroll
    for (int c = 0; c < 6; ++c) s += w1[o * 6 + c] * sfeat[c][kk];
    s = s * (g1[o] * rs) + be1[o];
    sh1[o][kk] = fmaxf(s, 0.0f);
  }
  __syncthreads();
  const int o = tid;  // 128 outputs, 128 threads
  const float sc = g11[o] * rs, bb = be11[o], bia = b11[o];
  float mx = -3.0e38f;
  for (int kk = 0; kk < K_LOC; ++kk) {
    float s = bia;
    for (int c = 0; c < 64; ++c) s += w11[o * 64 + c] * sh1[c][kk];
    mx = fmaxf(mx, fmaxf(s * sc + bb, 0.0f));
  }
  x1[((long)b * 128 + o) * N_PTS + n] = mx;
}

// ---------------------------------------------------------------------------
// Misc small kernels
// ---------------------------------------------------------------------------
__global__ void diag_kernel(const float* __restrict__ gram, float* __restrict__ xx) {
  const int t = blockIdx.x * blockDim.x + threadIdx.x;
  if (t >= B_SZ * N_PTS) return;
  const int b = t >> 10, n = t & 1023;
  xx[t] = gram[((long)b * N_PTS + n) * N_PTS + n];
}

__global__ void knn2_kernel(const float* __restrict__ gram, const float* __restrict__ xx,
                            int* __restrict__ idx) {
  const int t = blockIdx.x * blockDim.x + threadIdx.x;
  if (t >= B_SZ * N_PTS) return;
  const int b = t >> 10, n = t & 1023;
  const float* Gr = gram + ((long)b * N_PTS + n) * N_PTS;
  const float* xb = xx + b * N_PTS;
  const float xn = xb[n];
  float bd[K_GDM]; int bi[K_GDM];
  for (int i = 0; i < K_GDM; ++i) { bd[i] = 3.0e38f; bi[i] = 0; }
  for (int m = 0; m < N_PTS; ++m) {
    const float dsq = xn + xb[m] - 2.0f * Gr[m];
    if (dsq < bd[K_GDM - 1]) {
      int p = K_GDM - 1;
      while (p > 0 && bd[p - 1] > dsq) { bd[p] = bd[p - 1]; bi[p] = bi[p - 1]; --p; }
      bd[p] = dsq; bi[p] = m;
    }
  }
  int* o = idx + (long)t * K_GDM;
  for (int i = 0; i < K_GDM; ++i) o[i] = bi[i];
}

// Graph low-pass smoothing + variation score.  One block (128 thr) per (b,n).
__global__ void gdm_var_kernel(const float* __restrict__ x1, const int* __restrict__ idx2,
                               const float* __restrict__ gram, const float* __restrict__ xx,
                               float* __restrict__ var) {
  __shared__ float e[K_GDM];
  __shared__ int   nb[K_GDM];
  __shared__ float sdeg;
  __shared__ float red[128];
  const int t = blockIdx.x;
  const int b = t >> 10, n = t & 1023;
  const int tid = threadIdx.x;
  if (tid < K_GDM) {
    const int m = idx2[(long)t * K_GDM + tid];
    nb[tid] = m;
    float dsq = xx[b * N_PTS + n] + xx[b * N_PTS + m]
              - 2.0f * gram[((long)b * N_PTS + n) * N_PTS + m];
    dsq = fmaxf(dsq, 0.0f);
    const float dist = sqrtf(dsq + 1e-12f);
    const float w = expf(-dist * dist / 8.0f);   // sigma = 2
    e[tid] = (dist < 0.2f) ? w : 0.0f;           // tau = 0.2
  }
  __syncthreads();
  if (tid == 0) {
    float s = 0.0f;
    for (int i = 0; i < K_GDM; ++i) s += e[i];
    sdeg = s + 1e-9f;
  }
  __syncthreads();
  const float* xb = x1 + (long)b * 128 * N_PTS;
  {
    const int c = tid;  // 128 channels
    float sm = 0.0f;
    for (int kk = 0; kk < K_GDM; ++kk) sm += e[kk] * xb[(long)c * N_PTS + nb[kk]];
    sm /= sdeg;
    const float dd = xb[(long)c * N_PTS + n] - sm;
    red[c] = dd * dd;
  }
  __syncthreads();
  for (int s = 64; s > 0; s >>= 1) {
    if (tid < s) red[tid] += red[tid + s];
    __syncthreads();
  }
  if (tid == 0) var[b * N_PTS + n] = sqrtf(red[0] + 1e-12f);
}

// top-256 selection (sign=+1: largest var; sign=-1: smallest).  Block per batch.
__global__ void topm_kernel(const float* __restrict__ var, int* __restrict__ idx, float sign) {
  __shared__ float sv[N_PTS];
  __shared__ float bv[256];
  __shared__ int   bi2[256];
  const int b = blockIdx.x, tid = threadIdx.x;
  for (int i = tid; i < N_PTS; i += 256) sv[i] = sign * var[b * N_PTS + i];
  __syncthreads();
  for (int it = 0; it < M_GDM; ++it) {
    float best = -3.0e38f; int bidx = N_PTS;
    for (int i = tid; i < N_PTS; i += 256) {
      const float v = sv[i];
      if (v > best || (v == best && i < bidx)) { best = v; bidx = i; }
    }
    bv[tid] = best; bi2[tid] = bidx;
    __syncthreads();
    for (int s = 128; s > 0; s >>= 1) {
      if (tid < s) {
        if (bv[tid + s] > bv[tid] || (bv[tid + s] == bv[tid] && bi2[tid + s] < bi2[tid])) {
          bv[tid] = bv[tid + s]; bi2[tid] = bi2[tid + s];
        }
      }
      __syncthreads();
    }
    if (tid == 0) { idx[b * M_GDM + it] = bi2[0]; sv[bi2[0]] = -3.0e38f; }
    __syncthreads();
  }
}

__global__ void gather_cm_kernel(const float* __restrict__ x1, const int* __restrict__ idx,
                                 float* __restrict__ out) {
  const int t = blockIdx.x * blockDim.x + threadIdx.x;
  if (t >= B_SZ * 128 * M_GDM) return;
  const int m = t % M_GDM;
  const int c = (t / M_GDM) % 128;
  const int b = t / (M_GDM * 128);
  out[t] = x1[((long)b * 128 + c) * N_PTS + idx[b * M_GDM + m]];
}

__global__ void softmax_row_kernel(float* __restrict__ att) {
  __shared__ float red[256];
  const long r = blockIdx.x;
  const int tid = threadIdx.x;
  float* row = att + r * N_PTS;
  float mx = -3.0e38f;
  for (int i = tid; i < N_PTS; i += 256) mx = fmaxf(mx, row[i]);
  red[tid] = mx; __syncthreads();
  for (int s = 128; s > 0; s >>= 1) { if (tid < s) red[tid] = fmaxf(red[tid], red[tid + s]); __syncthreads(); }
  mx = red[0]; __syncthreads();
  float sum = 0.0f;
  for (int i = tid; i < N_PTS; i += 256) { const float ev = expf(row[i] - mx); row[i] = ev; sum += ev; }
  red[tid] = sum; __syncthreads();
  for (int s = 128; s > 0; s >>= 1) { if (tid < s) red[tid] += red[tid + s]; __syncthreads(); }
  const float inv = 1.0f / red[0];
  for (int i = tid; i < N_PTS; i += 256) row[i] *= inv;
}

// att[:,n,m] /= (1e-9 + sum_n att[:,n,m])
__global__ void colnorm_kernel(float* __restrict__ att) {
  const int t = blockIdx.x * blockDim.x + threadIdx.x;
  if (t >= B_SZ * N_PTS) return;
  const int b = t >> 10, m = t & 1023;
  float* base = att + (long)b * N_PTS * N_PTS + m;
  float s = 0.0f;
  for (int n = 0; n < N_PTS; ++n) s += base[(long)n * N_PTS];
  const float inv = 1.0f / (1e-9f + s);
  for (int n = 0; n < N_PTS; ++n) base[(long)n * N_PTS] *= inv;
}

// out = a(strided batch) - c(contiguous), (B,256,N)
__global__ void sub_kernel(const float* __restrict__ a, long aBs,
                           const float* __restrict__ c, float* __restrict__ out) {
  const long per = 256L * N_PTS;
  const long t = (long)blockIdx.x * blockDim.x + threadIdx.x;
  if (t >= (long)B_SZ * per) return;
  const long b = t / per, r = t - b * per;
  out[t] = a[b * aBs + r] - c[t];
}

__global__ void rowmax_kernel(const float* __restrict__ f, float* __restrict__ g) {
  const int t = blockIdx.x * blockDim.x + threadIdx.x;
  if (t >= B_SZ * 1024) return;
  const float* r = f + (long)t * N_PTS;
  float mx = r[0];
  for (int n = 1; n < N_PTS; ++n) mx = fmaxf(mx, r[n]);
  g[t] = mx;
}

// ---------------------------------------------------------------------------
// Host orchestration
// ---------------------------------------------------------------------------
static inline void gemm(hipStream_t st, int batch, int M, int N, int K,
                        const float* A, long Ars, long Acs, long Abs,
                        const float* Bp, long Brs, long Bcs, long Bbs,
                        float* C, long Crs, long Ccs, long Cbs,
                        const float* bias, const float* bng, const float* bnb,
                        const float* R, long Rrs, long Rcs, long Rbs,
                        int act, float alpha) {
  GemmDesc d;
  d.A = A; d.Ars = Ars; d.Acs = Acs; d.Abs = Abs;
  d.B = Bp; d.Brs = Brs; d.Bcs = Bcs; d.Bbs = Bbs;
  d.C = C; d.Crs = Crs; d.Ccs = Ccs; d.Cbs = Cbs;
  d.bias = bias; d.bng = bng; d.bnb = bnb;
  d.R = R; d.Rrs = Rrs; d.Rcs = Rcs; d.Rbs = Rbs;
  d.M = M; d.N = N; d.K = K; d.act = act; d.alpha = alpha;
  dim3 g((M + 15) / 16, (N + 63) / 64, batch);
  gemm_wmma_kernel<<<g, dim3(128), 0, st>>>(d);
}

static void run_sgcam(hipStream_t st, const float* x1, const float* x2, const float* const* p,
                      float* gbuf, float* thbuf, float* phbuf, float* Wbuf, float* ybuf,
                      float* outC, long outCbs) {
  const long Nn = N_PTS, Mm = M_GDM;
  const float *g_w = p[0], *g_b = p[1], *th_w = p[2], *th_b = p[3], *ph_w = p[4],
              *ph_b = p[5], *W_w = p[6], *W_b = p[7], *Wbn_g = p[8], *Wbn_b = p[9];
  // g = conv(x2) -> (B,64,M)
  gemm(st, B_SZ, 64, (int)Mm, 128, g_w, 128, 1, 0, x2, Mm, 1, 128 * Mm,
       gbuf, Mm, 1, 64 * Mm, g_b, nullptr, nullptr, nullptr, 0, 0, 0, 0, 1.0f);
  // th = conv(x1) -> (B,64,N)
  gemm(st, B_SZ, 64, (int)Nn, 128, th_w, 128, 1, 0, x1, Nn, 1, 128 * Nn,
       thbuf, Nn, 1, 64 * Nn, th_b, nullptr, nullptr, nullptr, 0, 0, 0, 0, 1.0f);
  // ph = conv(x2) -> (B,64,M)
  gemm(st, B_SZ, 64, (int)Mm, 128, ph_w, 128, 1, 0, x2, Mm, 1, 128 * Mm,
       phbuf, Mm, 1, 64 * Mm, ph_b, nullptr, nullptr, nullptr, 0, 0, 0, 0, 1.0f);
  // W_ = th^T * ph / M -> (B,N,M)
  gemm(st, B_SZ, (int)Nn, (int)Mm, 64, thbuf, 1, Nn, 64 * Nn, phbuf, Mm, 1, 64 * Mm,
       Wbuf, Mm, 1, Nn * Mm, nullptr, nullptr, nullptr, nullptr, 0, 0, 0, 0, 1.0f / (float)Mm);
  // y = g * W_^T -> (B,64,N)
  gemm(st, B_SZ, 64, (int)Nn, (int)Mm, gbuf, Mm, 1, 64 * Mm, Wbuf, 1, Mm, Nn * Mm,
       ybuf, Nn, 1, 64 * Nn, nullptr, nullptr, nullptr, nullptr, 0, 0, 0, 0, 1.0f);
  // out = bn(conv(y)) + x1 -> (B,128,N) written into fcat slice
  gemm(st, B_SZ, 128, (int)Nn, 64, W_w, 64, 1, 0, ybuf, Nn, 1, 64 * Nn,
       outC, Nn, 1, outCbs, W_b, Wbn_g, Wbn_b, x1, Nn, 1, 128 * Nn, 0, 1.0f);
}

static void run_sa(hipStream_t st, const float* in, long inBs, float* outC, long outCbs,
                   const float* const* p, float* kq, float* vbuf, float* att,
                   float* xr, float* dif) {
  const long Nn = N_PTS;
  const float *qk_w = p[0], *v_w = p[1], *v_b = p[2], *tr_w = p[3], *tr_b = p[4],
              *bn_g = p[5], *bn_b = p[6];
  // shared q/k projection (no bias) -> (B,64,N)
  gemm(st, B_SZ, 64, (int)Nn, 256, qk_w, 256, 1, 0, in, Nn, 1, inBs,
       kq, Nn, 1, 64 * Nn, nullptr, nullptr, nullptr, nullptr, 0, 0, 0, 0, 1.0f);
  // v -> (B,256,N)
  gemm(st, B_SZ, 256, (int)Nn, 256, v_w, 256, 1, 0, in, Nn, 1, inBs,
       vbuf, Nn, 1, 256 * Nn, v_b, nullptr, nullptr, nullptr, 0, 0, 0, 0, 1.0f);
  // att = kq^T * kq -> (B,N,N)
  gemm(st, B_SZ, (int)Nn, (int)Nn, 64, kq, 1, Nn, 64 * Nn, kq, Nn, 1, 64 * Nn,
       att, Nn, 1, Nn * Nn, nullptr, nullptr, nullptr, nullptr, 0, 0, 0, 0, 1.0f);
  softmax_row_kernel<<<B_SZ * N_PTS, 256, 0, st>>>(att);
  colnorm_kernel<<<(B_SZ * N_PTS + 255) / 256, 256, 0, st>>>(att);
  // x_r = v * att -> (B,256,N)
  gemm(st, B_SZ, 256, (int)Nn, (int)Nn, vbuf, Nn, 1, 256 * Nn, att, Nn, 1, Nn * Nn,
       xr, Nn, 1, 256 * Nn, nullptr, nullptr, nullptr, nullptr, 0, 0, 0, 0, 1.0f);
  // diff = in - x_r
  sub_kernel<<<(int)(((long)B_SZ * 256 * Nn + 255) / 256), 256, 0, st>>>(in, inBs, xr, dif);
  // out = relu(bn(conv(diff)+tr_b)) + in
  gemm(st, B_SZ, 256, (int)Nn, 256, tr_w, 256, 1, 0, dif, Nn, 1, 256 * Nn,
       outC, Nn, 1, outCbs, tr_b, bn_g, bn_b, in, Nn, 1, inBs, 1, 1.0f);
}

extern "C" void kernel_launch(void* const* d_in, const int* in_sizes, int n_in,
                              void* d_out, int out_size, void* d_ws, size_t ws_size,
                              hipStream_t stream) {
  (void)in_sizes; (void)n_in; (void)out_size; (void)ws_size;
  int ii = 0;
  auto F = [&](void) -> const float* { return (const float*)d_in[ii++]; };

  const float* x = F();
  ii++;  // normal: unused in this config
  const float* conv1_w = F();  const float* conv1_b = F();
  const float* bn1_g   = F();  const float* bn1_b   = F();
  const float* conv11_w = F(); const float* conv11_b = F();
  const float* bn11_g  = F();  const float* bn11_b  = F();
  const float* sgp[2][10];
  for (int s = 0; s < 2; ++s) for (int j = 0; j < 10; ++j) sgp[s][j] = F();
  const float* pt1_w = F(); const float* ptbn1_g = F(); const float* ptbn1_b = F();
  const float* pt2_w = F(); const float* ptbn2_g = F(); const float* ptbn2_b = F();
  const float* sap[4][7];
  for (int s = 0; s < 4; ++s) for (int j = 0; j < 7; ++j) sap[s][j] = F();
  const float* fuse_w = F(); const float* fuse_g = F(); const float* fuse_b = F();
  const float* lin1_w = F(); const float* bn6_g = F(); const float* bn6_b = F();
  const float* lin2_w = F(); const float* lin2_b = F();
  const float* bn7_g = F();  const float* bn7_b = F();
  const float* lin3_w = F(); const float* lin3_b = F();

  // ---- workspace bump allocator ----
  char* base = (char*)d_ws; size_t off = 0;
  auto alloc = [&](size_t bytes) -> void* {
    void* p = base + off; off += (bytes + 255) & ~(size_t)255; return p;
  };
  int*   idx1  = (int*)  alloc((size_t)B_SZ * N_PTS * K_LOC * 4);
  float* x1    = (float*)alloc((size_t)B_SZ * 128 * N_PTS * 4);
  float* bigNN = (float*)alloc((size_t)B_SZ * N_PTS * N_PTS * 4);   // Gram, then att
  float* xx    = (float*)alloc((size_t)B_SZ * N_PTS * 4);
  int*   idx2  = (int*)  alloc((size_t)B_SZ * N_PTS * K_GDM * 4);
  float* var   = (float*)alloc((size_t)B_SZ * N_PTS * 4);
  int*   idxs  = (int*)  alloc((size_t)B_SZ * M_GDM * 4);
  int*   idxg  = (int*)  alloc((size_t)B_SZ * M_GDM * 4);
  float* xs2   = (float*)alloc((size_t)B_SZ * 128 * M_GDM * 4);
  float* xg2   = (float*)alloc((size_t)B_SZ * 128 * M_GDM * 4);
  float* gbuf  = (float*)alloc((size_t)B_SZ * 64 * M_GDM * 4);
  float* thbuf = (float*)alloc((size_t)B_SZ * 64 * N_PTS * 4);
  float* phbuf = (float*)alloc((size_t)B_SZ * 64 * M_GDM * 4);
  float* Wbuf  = (float*)alloc((size_t)B_SZ * N_PTS * M_GDM * 4);
  float* ybuf  = (float*)alloc((size_t)B_SZ * 64 * N_PTS * 4);
  float* fcat  = (float*)alloc((size_t)B_SZ * 1280 * N_PTS * 4);    // [t1|t2|t3|t4|feat1]
  float* ta    = (float*)alloc((size_t)B_SZ * 256 * N_PTS * 4);
  float* tb    = (float*)alloc((size_t)B_SZ * 256 * N_PTS * 4);
  float* kq    = (float*)alloc((size_t)B_SZ * 64 * N_PTS * 4);
  float* vbuf  = (float*)alloc((size_t)B_SZ * 256 * N_PTS * 4);
  float* xr    = (float*)alloc((size_t)B_SZ * 256 * N_PTS * 4);
  float* dif   = (float*)alloc((size_t)B_SZ * 256 * N_PTS * 4);
  float* fbuf  = (float*)alloc((size_t)B_SZ * 1024 * N_PTS * 4);
  float* gvec  = (float*)alloc((size_t)B_SZ * 1024 * 4);
  float* hl1   = (float*)alloc((size_t)512 * 16 * 4);
  float* hl2   = (float*)alloc((size_t)256 * 16 * 4);

  const long Nn = N_PTS;

  // 1) KNN(xyz, 30) + local operator -> x1 (B,128,N)
  knn3_kernel<<<(B_SZ * N_PTS + 255) / 256, 256, 0, stream>>>(x, idx1);
  local_op_kernel<<<B_SZ * N_PTS, 128, 0, stream>>>(
      x, idx1, conv1_w, conv1_b, bn1_g, bn1_b, conv11_w, conv11_b, bn11_g, bn11_b, x1);

  // 2) GDM: Gram(x1) via WMMA, KNN(128-dim, 64), smoothing/variation, top/bottom-256
  gemm(stream, B_SZ, N_PTS, N_PTS, 128, x1, 1, Nn, 128 * Nn, x1, Nn, 1, 128 * Nn,
       bigNN, Nn, 1, Nn * Nn, nullptr, nullptr, nullptr, nullptr, 0, 0, 0, 0, 1.0f);
  diag_kernel<<<(B_SZ * N_PTS + 255) / 256, 256, 0, stream>>>(bigNN, xx);
  knn2_kernel<<<(B_SZ * N_PTS + 127) / 128, 128, 0, stream>>>(bigNN, xx, idx2);
  gdm_var_kernel<<<B_SZ * N_PTS, 128, 0, stream>>>(x1, idx2, bigNN, xx, var);
  topm_kernel<<<B_SZ, 256, 0, stream>>>(var, idxs, 1.0f);
  topm_kernel<<<B_SZ, 256, 0, stream>>>(var, idxg, -1.0f);
  gather_cm_kernel<<<(B_SZ * 128 * M_GDM + 255) / 256, 256, 0, stream>>>(x1, idxs, xs2);
  gather_cm_kernel<<<(B_SZ * 128 * M_GDM + 255) / 256, 256, 0, stream>>>(x1, idxg, xg2);

  // 3) SGCAM x2 -> feat1 rows [1024..1279] of fcat
  float* feat1 = fcat + 1024L * N_PTS;
  run_sgcam(stream, x1, xs2, sgp[0], gbuf, thbuf, phbuf, Wbuf, ybuf,
            feat1, 1280L * Nn);
  run_sgcam(stream, x1, xg2, sgp[1], gbuf, thbuf, phbuf, Wbuf, ybuf,
            feat1 + 128L * N_PTS, 1280L * Nn);

  // 4) point-transformer stem convs
  gemm(stream, B_SZ, 256, N_PTS, 256, pt1_w, 256, 1, 0, feat1, Nn, 1, 1280 * Nn,
       ta, Nn, 1, 256 * Nn, nullptr, ptbn1_g, ptbn1_b, nullptr, 0, 0, 0, 1, 1.0f);
  gemm(stream, B_SZ, 256, N_PTS, 256, pt2_w, 256, 1, 0, ta, Nn, 1, 256 * Nn,
       tb, Nn, 1, 256 * Nn, nullptr, ptbn2_g, ptbn2_b, nullptr, 0, 0, 0, 1, 1.0f);

  // 5) four SA layers, outputs written straight into the concat buffer
  const float* sain = tb; long sainBs = 256L * Nn;
  for (int l = 0; l < 4; ++l) {
    float* outp = fcat + (long)l * 256 * N_PTS;
    run_sa(stream, sain, sainBs, outp, 1280L * Nn, sap[l], kq, vbuf, bigNN, xr, dif);
    sain = outp; sainBs = 1280L * Nn;
  }

  // 6) fuse conv (1024x1280) + leaky + global max pool
  gemm(stream, B_SZ, 1024, N_PTS, 1280, fuse_w, 1280, 1, 0, fcat, Nn, 1, 1280 * Nn,
       fbuf, Nn, 1, 1024 * Nn, nullptr, fuse_g, fuse_b, nullptr, 0, 0, 0, 2, 1.0f);
  rowmax_kernel<<<(B_SZ * 1024 + 255) / 256, 256, 0, stream>>>(fbuf, gvec);

  // 7) MLP head (kept column-major as (C_out, B) until the final store)
  gemm(stream, 1, 512, 16, 1024, lin1_w, 1024, 1, 0, gvec, 1, 1024, 0,
       hl1, 16, 1, 0, nullptr, bn6_g, bn6_b, nullptr, 0, 0, 0, 2, 1.0f);
  gemm(stream, 1, 256, 16, 512, lin2_w, 512, 1, 0, hl1, 16, 1, 0,
       hl2, 16, 1, 0, lin2_b, bn7_g, bn7_b, nullptr, 0, 0, 0, 2, 1.0f);
  gemm(stream, 1, 40, 16, 256, lin3_w, 256, 1, 0, hl2, 16, 1, 0,
       (float*)d_out, 1, 40, 0, lin3_b, nullptr, nullptr, nullptr, 0, 0, 0, 0, 1.0f);
}
